// NextBytePredictor_17300128268701
// MI455X (gfx1250) — compile-verified, matrix operand order checked by reading, not run on hardware
//
#include <hip/hip_runtime.h>
#include <hip/hip_bf16.h>

typedef __bf16 bf16_t;
typedef __attribute__((ext_vector_type(16))) __bf16 v16bf;
typedef __attribute__((ext_vector_type(8)))  float  v8f;
typedef __attribute__((ext_vector_type(4)))  unsigned int u32x4;
typedef __attribute__((ext_vector_type(8)))  int i32x8;
typedef __attribute__((ext_vector_type(4)))  int i32x4;

#define B_SZ   8192
#define CTX    16
#define HID    512
#define IN_SZ  1472
#define K0     (IN_SZ + HID)      // 1984: [X | h0_l0]
#define K1     (HID + HID)        // 1024: [h1 | h0_l1]
#define NG     (4 * HID)          // 2048 gate columns
#define NOUT   256

#define TILE_ELEMS 5120           // 128 rows x 40 (32 data + 8 pad) bf16
#define TILE_BYTES 10240

#if defined(__has_builtin)
#if __has_builtin(__builtin_amdgcn_tensor_load_to_lds)
#define HAVE_TDM 1
#endif
#endif
#ifndef HAVE_TDM
#define HAVE_TDM 0
#endif

// ---------- helpers ----------
__device__ inline bf16_t f2bf(float f) {
  unsigned u = __builtin_bit_cast(unsigned, f);
  unsigned r = (u + 0x7FFFu + ((u >> 16) & 1u)) >> 16;
  unsigned short hs = (unsigned short)r;
  return __builtin_bit_cast(bf16_t, hs);
}
__device__ inline v16bf ld_v16bf(const bf16_t* p) {
  struct S { uint4 lo, hi; } s;
  s.lo = *(const uint4*)p;
  s.hi = *(const uint4*)(p + 8);
  return __builtin_bit_cast(v16bf, s);
}
__device__ inline float sigm(float x) { return 1.f / (1.f + __expf(-x)); }

#if HAVE_TDM
// Issue one TDM transfer: global [128 rows x 32 bf16], row stride K elems,
// into LDS at lds_off with 16B padding after each 64B row (row pitch 80B).
__device__ inline void tdm_load_tile(unsigned lds_off, const bf16_t* gptr, int K) {
  unsigned long long ga = (unsigned long long)(uintptr_t)gptr;
  u32x4 g0;
  g0[0] = 1u;                                            // count=1, load, user
  g0[1] = lds_off;                                       // lds_addr
  g0[2] = (unsigned)(ga & 0xFFFFFFFFu);                  // global_addr[31:0]
  g0[3] = (unsigned)((ga >> 32) & 0x01FFFFFFu) | (2u << 30); // [56:32] | type=2
  i32x8 g1;
  const unsigned td0 = 1u << 24;                         // tensor_dim0 (generous)
  const unsigned td1 = 1u << 20;                         // tensor_dim1 (generous)
  g1[0] = (int)((1u << 16)      // data_size = 2 bytes
              | (1u << 20)      // pad_enable
              | (3u << 22)      // pad_interval: 16 DWORDs (64B) between pads
              | (3u << 25));    // pad_amount: 4 DWORDs (16B) of padding
  g1[1] = (int)((td0 & 0xFFFFu) << 16);                  // atomic_addr=0 | td0 lo
  g1[2] = (int)(((td0 >> 16) & 0xFFFFu) | ((td1 & 0xFFFFu) << 16));
  g1[3] = (int)(((td1 >> 16) & 0xFFFFu) | (32u << 16));  // tile_dim0 = 32 elems
  g1[4] = (int)(128u);                                   // tile_dim1=128, tile_dim2=0
  g1[5] = (int)(unsigned)K;                              // tensor_dim0_stride lo
  g1[6] = 0;                                             // stride hi | dim1_stride lo
  g1[7] = 0;
  i32x4 z4 = {0, 0, 0, 0};
#if __clang_major__ >= 23
  i32x8 z8 = {0, 0, 0, 0, 0, 0, 0, 0};
  __builtin_amdgcn_tensor_load_to_lds(g0, g1, z4, z4, z8, 0);
#else
  __builtin_amdgcn_tensor_load_to_lds(g0, g1, z4, z4, 0);
#endif
}
#endif

// ---------- 1) embedding gather -> Xcat bf16 [B, 1984] ----------
__global__ __launch_bounds__(256) void embed_gather(
    const int* __restrict__ byte_vals, const int* __restrict__ ci0,
    const int* __restrict__ ci1, const int* __restrict__ ci2,
    const int* __restrict__ ci3, const float* __restrict__ numerical,
    const float* __restrict__ h0,  // layer-0 slice [B, HID]
    const float* __restrict__ bt, const float* __restrict__ t0,
    const float* __restrict__ t1, const float* __restrict__ t2,
    const float* __restrict__ t3, bf16_t* __restrict__ Xcat) {
  int idx = blockIdx.x * 256 + threadIdx.x;
  if (idx >= B_SZ * K0) return;
  int b = idx / K0, k = idx - b * K0;
  float v;
  if (k < IN_SZ) {
    int t = k / 92, off = k - t * 92;
    int bt_i = b * CTX + t;
    if      (off < 32) v = bt[byte_vals[bt_i] * 32 + off];
    else if (off < 45) v = t0[ci0[bt_i] * 13 + (off - 32)];
    else if (off < 51) v = t1[ci1[bt_i] * 6  + (off - 45)];
    else if (off < 55) v = t2[ci2[bt_i] * 4  + (off - 51)];
    else if (off < 84) v = t3[ci3[bt_i] * 29 + (off - 55)];
    else               v = numerical[bt_i * 8 + (off - 84)];
  } else {
    v = h0[(size_t)b * HID + (k - IN_SZ)];
  }
  Xcat[idx] = f2bf(v);
}

// ---------- 2) weight concat + fp32->bf16, keep [N, K] row-major ----------
__global__ __launch_bounds__(256) void concat_convert(
    const float* __restrict__ A, int Ka, const float* __restrict__ Bp, int Kb,
    bf16_t* __restrict__ dst, int N) {
  int K = Ka + Kb;
  int idx = blockIdx.x * 256 + threadIdx.x;
  if (idx >= N * K) return;
  int n = idx / K, k = idx - n * K;
  float v = (k < Ka) ? A[(size_t)n * Ka + k] : Bp[(size_t)n * Kb + (k - Ka)];
  dst[idx] = f2bf(v);
}

// ---------- 3) strided fp32 -> bf16 copy (h0[1] into X1 tail) ----------
__global__ __launch_bounds__(256) void copy_bf16_strided(
    const float* __restrict__ src, bf16_t* __restrict__ dst,
    int rows, int cols, int dstStride, int dstOff) {
  int idx = blockIdx.x * 256 + threadIdx.x;
  if (idx >= rows * cols) return;
  int r = idx / cols, c = idx - r * cols;
  dst[(size_t)r * dstStride + dstOff + c] = f2bf(src[idx]);
}

// ---------- 4) bf16 WMMA GEMM: C[M,N] = A[M,K] * Bw[N,K]^T (+bias) ----------
// BM=BN=128, BK=32, 256 threads = 8 wave32s; TDM double-buffered staging.
__device__ inline void compute_tile(const bf16_t* Asc, const bf16_t* Bsc,
                                    int waveM, int waveN, int l15, int half,
                                    v8f (&acc)[2][4]) {
  v16bf a[2], b[4];
#pragma unroll
  for (int i = 0; i < 2; ++i)
    a[i] = ld_v16bf(&Asc[(waveM * 32 + i * 16 + l15) * 40 + 16 * half]);
#pragma unroll
  for (int j = 0; j < 4; ++j)
    b[j] = ld_v16bf(&Bsc[(waveN * 64 + j * 16 + l15) * 40 + 16 * half]);
#pragma unroll
  for (int i = 0; i < 2; ++i)
#pragma unroll
    for (int j = 0; j < 4; ++j)
      acc[i][j] = __builtin_amdgcn_wmma_f32_16x16x32_bf16(
          false, a[i], false, b[j], (short)0, acc[i][j], false, false);
}

__global__ __launch_bounds__(256) void gemm_bf16_wmma(
    const bf16_t* __restrict__ A, const bf16_t* __restrict__ Bw,
    float* __restrict__ C, const float* __restrict__ bias,
    int M, int N, int K) {
  __shared__ __align__(16) bf16_t As[2 * TILE_ELEMS];
  __shared__ __align__(16) bf16_t Bs[2 * TILE_ELEMS];

  const int nTilesN = N >> 7;
  const int bm = blockIdx.x / nTilesN;
  const int bn = blockIdx.x - bm * nTilesN;
  const int tid  = threadIdx.x;
  const int lane = tid & 31;
  const int wave = tid >> 5;
  const int half = lane >> 4;   // K half (0:K 0..15, 1:K 16..31)
  const int l15  = lane & 15;   // M (A) / N (B) row within 16-tile
  const int waveM = wave >> 1;  // 0..3 -> 32 rows each
  const int waveN = wave & 1;   // 0..1 -> 64 cols each

  v8f acc[2][4];
#pragma unroll
  for (int i = 0; i < 2; ++i)
#pragma unroll
    for (int j = 0; j < 4; ++j) acc[i][j] = (v8f)(0.f);

  const bf16_t* Abase = A  + (size_t)(bm * 128) * K;
  const bf16_t* Bbase = Bw + (size_t)(bn * 128) * K;

#if HAVE_TDM
  const unsigned asLds = (unsigned)(uintptr_t)As;
  const unsigned bsLds = (unsigned)(uintptr_t)Bs;
  const int KT = K >> 5;
  if (wave == 0) {  // wave 0 drives the Tensor Data Mover
    tdm_load_tile(asLds, Abase, K);
    tdm_load_tile(bsLds, Bbase, K);
  }
  for (int t = 0; t < KT; ++t) {
    const int cur = t & 1, nxt = cur ^ 1;
    __syncthreads();  // buf[nxt] fully consumed by previous iteration
    if (wave == 0) {
      if (t + 1 < KT) {
        tdm_load_tile(asLds + (unsigned)nxt * TILE_BYTES, Abase + (t + 1) * 32, K);
        tdm_load_tile(bsLds + (unsigned)nxt * TILE_BYTES, Bbase + (t + 1) * 32, K);
        __builtin_amdgcn_s_wait_tensorcnt(2);  // cur pair landed, nxt in flight
      } else {
        __builtin_amdgcn_s_wait_tensorcnt(0);
      }
    }
    __syncthreads();  // cur tile visible to all waves
    compute_tile(As + cur * TILE_ELEMS, Bs + cur * TILE_ELEMS,
                 waveM, waveN, l15, half, acc);
  }
#else
  for (int k0 = 0; k0 < K; k0 += 32) {
    // batched staging: issue all global loads, then all LDS stores
    int row0 = tid >> 1, seg0 = (tid & 1) * 2;       // 2 x 16B per thread/matrix
    uint4 a0 = *(const uint4*)(Abase + (size_t)row0 * K + k0 + seg0 * 8);
    uint4 a1 = *(const uint4*)(Abase + (size_t)row0 * K + k0 + (seg0 + 1) * 8);
    uint4 b0 = *(const uint4*)(Bbase + (size_t)row0 * K + k0 + seg0 * 8);
    uint4 b1 = *(const uint4*)(Bbase + (size_t)row0 * K + k0 + (seg0 + 1) * 8);
    *(uint4*)&As[row0 * 40 + seg0 * 8]       = a0;
    *(uint4*)&As[row0 * 40 + (seg0 + 1) * 8] = a1;
    *(uint4*)&Bs[row0 * 40 + seg0 * 8]       = b0;
    *(uint4*)&Bs[row0 * 40 + (seg0 + 1) * 8] = b1;
    if (k0 + 32 < K) {
      __builtin_prefetch(Abase + (size_t)row0 * K + k0 + 32, 0, 1);
      __builtin_prefetch(Bbase + (size_t)row0 * K + k0 + 32, 0, 1);
    }
    __syncthreads();
    compute_tile(As, Bs, waveM, waveN, l15, half, acc);
    __syncthreads();
  }
#endif

  // epilogue: lane l holds rows m = r + 8*half (r=0..7), col = l15
#pragma unroll
  for (int i = 0; i < 2; ++i) {
#pragma unroll
    for (int j = 0; j < 4; ++j) {
      int mBase = bm * 128 + waveM * 32 + i * 16 + half * 8;
      int nCol  = bn * 128 + waveN * 64 + j * 16 + l15;
      float bv = bias ? bias[nCol] : 0.f;
#pragma unroll
      for (int r = 0; r < 8; ++r)
        C[(size_t)(mBase + r) * N + nCol] = acc[i][j][r] + bv;
    }
  }
}

// ---------- 5) LSTM pointwise: gates -> h, c (+ bf16 h for next GEMM) ------
__global__ __launch_bounds__(256) void lstm_pointwise(
    const float* __restrict__ gates, const float* __restrict__ b_ih,
    const float* __restrict__ b_hh, const float* __restrict__ c_prev,
    float* __restrict__ h_out, float* __restrict__ c_out,
    bf16_t* __restrict__ h_bf16, int ld_bf16) {
  int idx = blockIdx.x * 256 + threadIdx.x;
  if (idx >= B_SZ * HID) return;
  int b = idx >> 9, u = idx & (HID - 1);
  const float* g = gates + (size_t)b * NG;
  float gi = g[u]            + b_ih[u]            + b_hh[u];
  float gf = g[HID + u]      + b_ih[HID + u]      + b_hh[HID + u];
  float gg = g[2 * HID + u]  + b_ih[2 * HID + u]  + b_hh[2 * HID + u];
  float go = g[3 * HID + u]  + b_ih[3 * HID + u]  + b_hh[3 * HID + u];
  float c = sigm(gf) * c_prev[idx] + sigm(gi) * tanhf(gg);
  float h = sigm(go) * tanhf(c);
  c_out[idx] = c;
  h_out[idx] = h;
  h_bf16[(size_t)b * ld_bf16 + u] = f2bf(h);
}

// ---------- launch ----------
extern "C" void kernel_launch(void* const* d_in, const int* in_sizes, int n_in,
                              void* d_out, int out_size, void* d_ws, size_t ws_size,
                              hipStream_t stream) {
  const int*   byte_vals = (const int*)d_in[0];
  const int*   ci0 = (const int*)d_in[1];
  const int*   ci1 = (const int*)d_in[2];
  const int*   ci2 = (const int*)d_in[3];
  const int*   ci3 = (const int*)d_in[4];
  const float* numerical = (const float*)d_in[5];
  const float* h0 = (const float*)d_in[6];   // [2, B, HID]
  const float* c0 = (const float*)d_in[7];   // [2, B, HID]
  const float* bt = (const float*)d_in[8];
  const float* t0 = (const float*)d_in[9];
  const float* t1 = (const float*)d_in[10];
  const float* t2 = (const float*)d_in[11];
  const float* t3 = (const float*)d_in[12];
  const float* W_ih0 = (const float*)d_in[13];
  const float* W_hh0 = (const float*)d_in[14];
  const float* b_ih0 = (const float*)d_in[15];
  const float* b_hh0 = (const float*)d_in[16];
  const float* W_ih1 = (const float*)d_in[17];
  const float* W_hh1 = (const float*)d_in[18];
  const float* b_ih1 = (const float*)d_in[19];
  const float* b_hh1 = (const float*)d_in[20];
  const float* W_out = (const float*)d_in[21];
  const float* b_out = (const float*)d_in[22];
  float* out = (float*)d_out;

  // workspace layout (bytes)
  char* ws = (char*)d_ws;
  bf16_t* Xcat  = (bf16_t*)(ws);                               // B x 1984
  bf16_t* Wcat0 = (bf16_t*)(ws + 32505856);                    // 2048 x 1984
  bf16_t* X1    = (bf16_t*)(ws + 40632320);                    // B x 1024
  bf16_t* Wcat1 = (bf16_t*)(ws + 57409536);                    // 2048 x 1024
  bf16_t* H2    = (bf16_t*)(ws + 61603840);                    // B x 512
  bf16_t* Wo    = (bf16_t*)(ws + 69992448);                    // 256 x 512
  float*  gates = (float*) (ws + 70254592);                    // B x 2048

  // d_out layout: logits[B,256] | h1 | h2 | c1 | c2 (each B*HID)
  float* logits = out;
  float* h1_out = out + 2097152;
  float* h2_out = out + 6291456;
  float* c1_out = out + 10485760;
  float* c2_out = out + 14680064;

  const float* h0_l0 = h0;
  const float* h0_l1 = h0 + (size_t)B_SZ * HID;
  const float* c0_l0 = c0;
  const float* c0_l1 = c0 + (size_t)B_SZ * HID;

  // 1) gather embeddings + h0[0] -> Xcat (bf16)
  embed_gather<<<(B_SZ * K0 + 255) / 256, 256, 0, stream>>>(
      byte_vals, ci0, ci1, ci2, ci3, numerical, h0_l0, bt, t0, t1, t2, t3, Xcat);
  // 2) h0[1] -> X1 tail
  copy_bf16_strided<<<(B_SZ * HID + 255) / 256, 256, 0, stream>>>(
      h0_l1, X1, B_SZ, HID, K1, HID);
  // 3) weights -> bf16 concat [N,K]
  concat_convert<<<(NG * K0 + 255) / 256, 256, 0, stream>>>(W_ih0, IN_SZ, W_hh0, HID, Wcat0, NG);
  concat_convert<<<(NG * K1 + 255) / 256, 256, 0, stream>>>(W_ih1, HID, W_hh1, HID, Wcat1, NG);
  concat_convert<<<(NOUT * HID + 255) / 256, 256, 0, stream>>>(W_out, HID, nullptr, 0, Wo, NOUT);

  // 4) layer 0: gates = Xcat @ Wcat0^T
  gemm_bf16_wmma<<<(B_SZ / 128) * (NG / 128), 256, 0, stream>>>(
      Xcat, Wcat0, gates, nullptr, B_SZ, NG, K0);
  lstm_pointwise<<<(B_SZ * HID + 255) / 256, 256, 0, stream>>>(
      gates, b_ih0, b_hh0, c0_l0, h1_out, c1_out, X1, K1);

  // 5) layer 1: gates = [h1|h0_l1] @ Wcat1^T
  gemm_bf16_wmma<<<(B_SZ / 128) * (NG / 128), 256, 0, stream>>>(
      X1, Wcat1, gates, nullptr, B_SZ, NG, K1);
  lstm_pointwise<<<(B_SZ * HID + 255) / 256, 256, 0, stream>>>(
      gates, b_ih1, b_hh1, c0_l1, h2_out, c2_out, H2, HID);

  // 6) logits = h2 @ W_out^T + b_out
  gemm_bf16_wmma<<<(B_SZ / 128) * (NOUT / 128), 256, 0, stream>>>(
      H2, Wo, logits, b_out, B_SZ, NOUT, HID);
}
